// IntraBagAttention_17532056502568
// MI455X (gfx1250) — matrix operand order repaired and checked by reading, not compiled, not run on hardware
//
#include <hip/hip_runtime.h>
#include <hip/hip_bf16.h>
#include <math.h>

typedef __bf16 bf16;
typedef __attribute__((ext_vector_type(16))) __bf16 v16bf;
typedef __attribute__((ext_vector_type(8)))  __bf16 bf16x8;
typedef __attribute__((ext_vector_type(4)))  __bf16 bf16x4;
typedef __attribute__((ext_vector_type(8)))  float   v8f;
typedef __attribute__((ext_vector_type(4)))  float   f32x4;

// Problem constants (from reference setup)
constexpr int N = 128;   // relations
constexpr int D = 768;   // feature dim
constexpr int M = 32;    // max bag size (padded)

// LDS layout (bytes). bag & attT are dead before full is written -> overlap.
constexpr int BAG_STRIDE  = 776;  // bf16 elems per row (768 + 8 pad, 16B-aligned rows)
constexpr int ATT_STRIDE  = 132;  // f32 elems per row of att^T (32 x 132)
constexpr int SM_STRIDE   = 40;   // bf16 elems per row of sm / attB (128 x 40)
constexpr int FULL_STRIDE = 132;  // f32 elems per row of full (128 x 132)

constexpr int OFF_BAG  = 0;                    // 32*776*2  = 49664
constexpr int OFF_ATT  = 49664;                // 32*132*4  = 16896 -> 66560
constexpr int OFF_FULL = 0;                    // 128*132*4 = 67584 (overlaps bag+attT)
constexpr int OFF_ATTB = 67584;                // 128*40*2  = 10240 -> 77824
constexpr int OFF_SM   = 77824;                // 10240     -> 88064
constexpr int SMEM_BYTES = 88064;

// ---- Kernel 0: convert fc_weight (N x D f32) -> bf16 in workspace ----
__global__ void wconv_kernel(const float* __restrict__ w, bf16* __restrict__ wb, int n4) {
    int i = blockIdx.x * blockDim.x + threadIdx.x;
    if (i < n4) {
        f32x4 v = ((const f32x4*)w)[i];
        bf16x4 o;
#pragma unroll
        for (int j = 0; j < 4; ++j) o[j] = (bf16)v[j];
        ((bf16x4*)wb)[i] = o;
    }
}

// ---- Kernel 1: one workgroup per bag ----
__launch_bounds__(256)
__global__ void bagatt_kernel(const float* __restrict__ rep,
                              const bf16*  __restrict__ wb,     // N x D bf16 (row-major)
                              const float* __restrict__ bias,   // N
                              const int*   __restrict__ scope,  // B x 2
                              float*       __restrict__ out)    // B x N
{
    extern __shared__ char smem[];
    bf16*  bag  = (bf16*) (smem + OFF_BAG);
    float* attT = (float*)(smem + OFF_ATT);
    float* full = (float*)(smem + OFF_FULL);
    bf16*  attB = (bf16*) (smem + OFF_ATTB);
    bf16*  smx  = (bf16*) (smem + OFF_SM);

    const int b   = blockIdx.x;
    const int tid = threadIdx.x;
    const int start = scope[2 * b];
    int size = scope[2 * b + 1] - start;
    if (size > M) size = M;
    if (size < 0) size = 0;

    // ---- Phase 1: stage bag rows into LDS as bf16, zero-fill rows >= size ----
    for (int i = tid; i < (M * D) / 4; i += 256) {
        const int e   = i * 4;
        const int row = e / D;
        const int col = e - row * D;
        f32x4 v = {0.f, 0.f, 0.f, 0.f};
        if (row < size)
            v = *(const f32x4*)(rep + (size_t)(start + row) * D + col);
        bf16x4 o;
#pragma unroll
        for (int j = 0; j < 4; ++j) o[j] = (bf16)v[j];
        *(bf16x4*)(bag + row * BAG_STRIDE + col) = o;
    }
    __syncthreads();

    const int wave = tid >> 5;      // 0..7
    const int lane = tid & 31;
    const int l16  = lane & 15;
    const int half = lane >> 4;     // 0/1

    union AB { v16bf v; bf16x8 h[2]; };

    // ---- Phase 2: att^T (32 x 128) = bag (32xD) * W^T, bf16 WMMA, f32 acc ----
    // wave w owns n-tile w (cols 16w..16w+15), m-tiles 0 and 1.
    {
        v8f acc0 = {};
        v8f acc1 = {};
        const int  nbase = wave * 16;
        const bf16* wrow = wb + (size_t)(nbase + l16) * D;  // B col = W row
#pragma unroll 4
        for (int kt = 0; kt < D / 32; ++kt) {
            const int k0 = kt * 32;
            AB a0, a1, bb;
            // A layout: lanes 0-15 row=l16, K0..7 (VGPR0-3) & K16..23 (VGPR4-7);
            //           lanes 16-31 same row, K8..15 & K24..31.
            const bf16* pa0 = bag + l16 * BAG_STRIDE + k0 + half * 8;
            a0.h[0] = *(const bf16x8*)(pa0);
            a0.h[1] = *(const bf16x8*)(pa0 + 16);
            const bf16* pa1 = pa0 + 16 * BAG_STRIDE;
            a1.h[0] = *(const bf16x8*)(pa1);
            a1.h[1] = *(const bf16x8*)(pa1 + 16);
            // B layout: lanes 0-15 col=l16 hold K0..7 & K16..23; lanes 16-31 hold K8..15 & K24..31.
            const bf16* pb = wrow + k0 + half * 8;
            bb.h[0] = *(const bf16x8*)(pb);
            bb.h[1] = *(const bf16x8*)(pb + 16);
            acc0 = __builtin_amdgcn_wmma_f32_16x16x32_bf16(false, a0.v, false, bb.v,
                                                           (short)0, acc0, false, false);
            acc1 = __builtin_amdgcn_wmma_f32_16x16x32_bf16(false, a1.v, false, bb.v,
                                                           (short)0, acc1, false, false);
        }
        // D layout: VGPR r -> row r (lanes 0-15) / row 8+r (lanes 16-31), col l16.
#pragma unroll
        for (int r = 0; r < 8; ++r) {
            attT[(half * 8 + r) * ATT_STRIDE + nbase + l16]        = acc0[r];
            attT[(16 + half * 8 + r) * ATT_STRIDE + nbase + l16]   = acc1[r];
        }
    }
    __syncthreads();

    // ---- Phase 3: masked softmax over m per relation n; emit bf16 sm & attB ----
    if (tid < N) {
        const int n = tid;
        float mx = -INFINITY;
        for (int m = 0; m < size; ++m)
            mx = fmaxf(mx, attT[m * ATT_STRIDE + n]);
        float sum = 0.f;
        for (int m = 0; m < size; ++m)
            sum += __expf(attT[m * ATT_STRIDE + n] - mx);
        const float inv = 1.f / sum;
#pragma unroll
        for (int m = 0; m < M; ++m) {
            const bool ok = (m < size);
            const float a = ok ? attT[m * ATT_STRIDE + n] : 0.f;
            const float s = ok ? __expf(a - mx) * inv : 0.f;
            smx [n * SM_STRIDE + m] = (bf16)s;   // A for full: row n, K=m
            attB[n * SM_STRIDE + m] = (bf16)a;   // B for full: col n, K=m (finite!)
        }
    }
    __syncthreads();

    // ---- Phase 4: full (128x128) = sm (128x32) * att^T (32x128) + bias, K=32 ----
    {
        const int mbase = wave * 16;
        AB aSM;
        const bf16* ps = smx + (mbase + l16) * SM_STRIDE + half * 8;
        aSM.h[0] = *(const bf16x8*)(ps);
        aSM.h[1] = *(const bf16x8*)(ps + 16);
#pragma unroll
        for (int ct = 0; ct < 8; ++ct) {
            const int cb = ct * 16;
            AB bb;
            const bf16* pb = attB + (cb + l16) * SM_STRIDE + half * 8;
            bb.h[0] = *(const bf16x8*)(pb);
            bb.h[1] = *(const bf16x8*)(pb + 16);
            v8f acc = {};
            acc = __builtin_amdgcn_wmma_f32_16x16x32_bf16(false, aSM.v, false, bb.v,
                                                          (short)0, acc, false, false);
            const float bv = bias[cb + l16];
#pragma unroll
            for (int r = 0; r < 8; ++r)
                full[(mbase + half * 8 + r) * FULL_STRIDE + cb + l16] = acc[r] + bv;
        }
    }
    __syncthreads();

    // ---- Phase 5: softmax over k per row n, output diagonal ----
    if (tid < N) {
        const int n = tid;
        const float* row = full + n * FULL_STRIDE;
        float mx = -INFINITY;
        for (int k = 0; k < N; ++k) mx = fmaxf(mx, row[k]);
        float sum = 0.f;
        for (int k = 0; k < N; ++k) sum += __expf(row[k] - mx);
        out[(size_t)b * N + n] = __expf(row[n] - mx) / sum;
    }
}

extern "C" void kernel_launch(void* const* d_in, const int* in_sizes, int n_in,
                              void* d_out, int out_size, void* d_ws, size_t ws_size,
                              hipStream_t stream) {
    const float* rep   = (const float*)d_in[0];
    const float* w     = (const float*)d_in[1];
    const float* bias  = (const float*)d_in[2];
    const int*   scope = (const int*)d_in[3];
    const int B = in_sizes[3] / 2;

    bf16* wb = (bf16*)d_ws;  // N*D bf16 = 192 KB in workspace
    const int n4 = (N * D) / 4;
    wconv_kernel<<<(n4 + 255) / 256, 256, 0, stream>>>(w, wb, n4);
    bagatt_kernel<<<B, 256, SMEM_BYTES, stream>>>(rep, wb, bias, scope, (float*)d_out);
}
